// FocalSelfAttention_32074815766836
// MI455X (gfx1250) — matrix operand order repaired and checked by reading, not compiled, not run on hardware
//
#include <hip/hip_runtime.h>
#include <hip/hip_bf16.h>
#include <math.h>

// ---------------------------------------------------------------------------
// FocalSelfAttention for MI455X (gfx1250, wave32, WMMA 16x16x32 f16->f32)
// B=2, C=256, H=W=64, heads=8, d=32, win=4 -> nW=256, Tq=16
// local keys Ll=64, mid keys Lm=1024, global keys Lg=256  (1344 keys/query)
// ---------------------------------------------------------------------------

typedef __attribute__((ext_vector_type(16))) _Float16 v16h;
typedef __attribute__((ext_vector_type(8)))  float    v8f;

#define C_DIM 256
#define NWIN  256
#define HEADS 8
#define HD    32
#define HW    4096   // H*W

// ---- fragment loaders (k = 16*(lane/16) + s, contiguous per lane) ----------------------
__device__ __forceinline__ v16h ld2x16B(const _Float16* p0, const _Float16* p1) {
  union { uint4 u[2]; v16h v; } t;
  t.u[0] = *reinterpret_cast<const uint4*>(p0);
  t.u[1] = *reinterpret_cast<const uint4*>(p1);
  return t.v;
}
__device__ __forceinline__ v16h ld32B(const _Float16* p) {
  union { uint4 u[2]; v16h v; } t;
  const uint4* q = reinterpret_cast<const uint4*>(p);
  t.u[0] = q[0]; t.u[1] = q[1];
  return t.v;
}
__device__ __forceinline__ v8f wmma16(v16h a, v16h b, v8f c) {
  // D = A(16x32 f16) * B(32x16 f16) + C(16x16 f32)
  return __builtin_amdgcn_wmma_f32_16x16x32_f16(false, a, false, b, (short)0, c,
                                                false, false);
}

// ---------------------------------------------------------------------------
// Stage 0: weights -> f16, transposed to [N][K] for contiguous B-fragments
// ---------------------------------------------------------------------------
__global__ __launch_bounds__(256) void convert_weights(
    const float* __restrict__ qkv_w, const float* __restrict__ out_w,
    _Float16* __restrict__ wqt, _Float16* __restrict__ wkt,
    _Float16* __restrict__ wvt, _Float16* __restrict__ wot) {
  int t = blockIdx.x * 256 + threadIdx.x;   // t = n*256 + k
  int n = t >> 8, k = t & 255;
  wqt[t] = (_Float16)qkv_w[k * 768 + n];
  wkt[t] = (_Float16)qkv_w[k * 768 + 256 + n];
  wvt[t] = (_Float16)qkv_w[k * 768 + 512 + n];
  wot[t] = (_Float16)out_w[k * 256 + n];
}

// ---------------------------------------------------------------------------
// Stage 1: token builders (gather + LayerNorm / pooling) -> f16 [token][256]
// ---------------------------------------------------------------------------
__global__ __launch_bounds__(256) void build_qtok(
    const float* __restrict__ feat, const float* __restrict__ lnw,
    const float* __restrict__ lnb, _Float16* __restrict__ qtok) {
  int t = blockIdx.x;               // 0..8191 : b*4096 + w*16 + q
  int c = threadIdx.x;
  int b = t >> 12, w = (t >> 4) & 255, q = t & 15;
  int y = (w >> 4) * 4 + (q >> 2);
  int x = (w & 15) * 4 + (q & 3);
  float v = feat[((size_t)(b * C_DIM + c)) * HW + y * 64 + x];
  __shared__ float s1[256], s2[256];
  s1[c] = v; s2[c] = v * v;
  __syncthreads();
  for (int off = 128; off > 0; off >>= 1) {
    if (c < off) { s1[c] += s1[c + off]; s2[c] += s2[c + off]; }
    __syncthreads();
  }
  float mu  = s1[0] * (1.0f / 256.0f);
  float var = s2[0] * (1.0f / 256.0f) - mu * mu;
  float nv  = (v - mu) * rsqrtf(var + 1e-5f) * lnw[c] + lnb[c];
  qtok[(size_t)t * 256 + c] = (_Float16)nv;
}

__global__ __launch_bounds__(256) void build_loc(
    const float* __restrict__ feat, _Float16* __restrict__ loc) {
  int t = blockIdx.x;               // 0..32767 : b*16384 + w*64 + l
  int c = threadIdx.x;
  int b = t >> 14, w = (t >> 6) & 255, l = t & 63;
  int y = (w >> 4) * 4 - 2 + (l >> 3);
  int x = (w & 15) * 4 - 2 + (l & 7);
  float v = 0.0f;
  if (y >= 0 && y < 64 && x >= 0 && x < 64)
    v = feat[((size_t)(b * C_DIM + c)) * HW + y * 64 + x];
  loc[(size_t)t * 256 + c] = (_Float16)v;
}

__global__ __launch_bounds__(256) void build_mid(
    const float* __restrict__ feat, _Float16* __restrict__ mid) {
  int t = blockIdx.x;               // 0..2047 : b*1024 + my*32 + mx
  int c = threadIdx.x;
  int b = t >> 10, m = t & 1023;
  int my = m >> 5, mx = m & 31;
  const float* f = feat + ((size_t)(b * C_DIM + c)) * HW;
  float v = 0.25f * (f[(2*my)*64 + 2*mx]     + f[(2*my)*64 + 2*mx + 1] +
                     f[(2*my+1)*64 + 2*mx]   + f[(2*my+1)*64 + 2*mx + 1]);
  mid[(size_t)t * 256 + c] = (_Float16)v;
}

__global__ __launch_bounds__(256) void build_glb(
    const float* __restrict__ feat, _Float16* __restrict__ glb) {
  int t = blockIdx.x;               // 0..511 : b*256 + gy*16 + gx
  int c = threadIdx.x;
  int b = t >> 8, m = t & 255;
  int gy = m >> 4, gx = m & 15;
  const float* f = feat + ((size_t)(b * C_DIM + c)) * HW;
  float v = 0.0f;
  for (int dy = 0; dy < 4; ++dy)
    for (int dx = 0; dx < 4; ++dx)
      v += f[(4*gy + dy) * 64 + 4*gx + dx];
  glb[(size_t)t * 256 + c] = (_Float16)(v * (1.0f / 16.0f));
}

// ---------------------------------------------------------------------------
// Stage 2: generic GEMM  Out[M,256] = A[M,256] * Bt^T + bias  (f16 in, f16 out)
// block = 128 threads (4 waves); wave -> 32x32 tile (2x2 WMMA frags)
// grid = (M/64, 256/64)
// ---------------------------------------------------------------------------
__global__ __launch_bounds__(128) void gemm_f16(
    const _Float16* __restrict__ A, const _Float16* __restrict__ Bt,
    const float* __restrict__ bias, _Float16* __restrict__ Out) {
  int wave = threadIdx.x >> 5;
  int lane = threadIdx.x & 31;
  int ll = lane & 15, g = lane >> 4;
  int rowbase = blockIdx.x * 64 + (wave >> 1) * 32;
  int colbase = blockIdx.y * 64 + (wave & 1) * 32;
  v8f acc[2][2] = {};
  for (int kb = 0; kb < 256; kb += 32) {
    v16h a[2], bb[2];
#pragma unroll
    for (int i = 0; i < 2; ++i) {
      const _Float16* ap = A + (size_t)(rowbase + i * 16 + ll) * 256 + kb + (g ? 16 : 0);
      a[i] = ld32B(ap);
      __builtin_prefetch(ap + 32, 0, 0);
      const _Float16* bp = Bt + (size_t)(colbase + i * 16 + ll) * 256 + kb + (g ? 16 : 0);
      bb[i] = ld32B(bp);
    }
#pragma unroll
    for (int i = 0; i < 2; ++i)
#pragma unroll
      for (int j = 0; j < 2; ++j)
        acc[i][j] = wmma16(a[i], bb[j], acc[i][j]);
  }
#pragma unroll
  for (int i = 0; i < 2; ++i)
#pragma unroll
    for (int j = 0; j < 2; ++j)
#pragma unroll
      for (int r = 0; r < 8; ++r) {
        int row = rowbase + i * 16 + r + 8 * g;
        int col = colbase + j * 16 + ll;
        Out[(size_t)row * 256 + col] = (_Float16)(acc[i][j][r] + bias[col]);
      }
}

// ---------------------------------------------------------------------------
// Stage 3: V transposes: [token][C] -> per-(b,[w,]head) [d=32][token]
// ---------------------------------------------------------------------------
__global__ __launch_bounds__(256) void transpose_local(
    const _Float16* __restrict__ V, _Float16* __restrict__ Vt) {
  int tid = blockIdx.x * 256 + threadIdx.x;   // 2*256*64*256 elements
  int c = tid & 255, tk = tid >> 8;
  int tok = tk & 63, w = (tk >> 6) & 255, b = tk >> 14;
  int h = c >> 5, dd = c & 31;
  Vt[(((size_t)(b * NWIN + w) * 8 + h) * 32 + dd) * 64 + tok] =
      V[(size_t)tk * 256 + c];
}

__global__ __launch_bounds__(256) void transpose_bl(
    const _Float16* __restrict__ V, _Float16* __restrict__ Vt, int L) {
  int tid = blockIdx.x * 256 + threadIdx.x;   // 2*L*256 elements
  int c = tid & 255, tk = tid >> 8;
  int b = tk / L, tok = tk - b * L;
  int h = c >> 5, dd = c & 31;
  Vt[(((size_t)(b * 8 + h) * 32 + dd)) * (size_t)L + tok] =
      V[(size_t)tk * 256 + c];
}

// ---------------------------------------------------------------------------
// Stage 4: attention. One wave per (b, window, head). Online softmax over
// 42 chunks of 32 keys (2 local + 32 mid + 8 global). 4 WMMAs per chunk.
// ---------------------------------------------------------------------------
__global__ __launch_bounds__(32) void attn_kernel(
    const _Float16* __restrict__ Q16,
    const _Float16* __restrict__ Kl, const _Float16* __restrict__ Km,
    const _Float16* __restrict__ Kg,
    const _Float16* __restrict__ Vlt, const _Float16* __restrict__ Vmt,
    const _Float16* __restrict__ Vgt,
    _Float16* __restrict__ Ob) {
  int blk = blockIdx.x;               // b*2048 + w*8 + head
  int head = blk & 7;
  int w = (blk >> 3) & 255;
  int b = blk >> 11;
  int lane = threadIdx.x;
  int ll = lane & 15, g = lane >> 4;

  __shared__ __align__(16) float    sS[16 * 32];
  __shared__ __align__(16) _Float16 sP[16 * 32];
  __shared__ float rowm[16], rowsum[16], rowsc[16];

  // Q fragment: 16 queries x d=32, reused for all chunks
  const _Float16* qp =
      Q16 + ((size_t)((b * NWIN + w) * 16 + ll)) * 256 + head * HD;
  v16h aq = ld32B(qp + (g ? 16 : 0));

  v8f acc0 = {}, acc1 = {};
  if (lane < 16) { rowm[lane] = -3.0e38f; rowsum[lane] = 0.0f; }
  __syncthreads();

  const float scale = 0.17677669529663687f;   // 1/sqrt(32)

  for (int ch = 0; ch < 42; ++ch) {
    const _Float16* kbase;
    const _Float16* vt;
    int vstride, tokoff;
    if (ch < 2) {
      kbase  = Kl + ((size_t)((b * NWIN + w) * 64 + ch * 32)) * 256 + head * HD;
      vt     = Vlt + ((size_t)((b * NWIN + w) * 8 + head)) * (32 * 64);
      vstride = 64;  tokoff = ch * 32;
    } else if (ch < 34) {
      int c2 = ch - 2;
      kbase  = Km + ((size_t)(b * 1024 + c2 * 32)) * 256 + head * HD;
      vt     = Vmt + ((size_t)(b * 8 + head)) * (32 * 1024);
      vstride = 1024; tokoff = c2 * 32;
    } else {
      int c3 = ch - 34;
      kbase  = Kg + ((size_t)(b * 256 + c3 * 32)) * 256 + head * HD;
      vt     = Vgt + ((size_t)(b * 8 + head)) * (32 * 256);
      vstride = 256; tokoff = c3 * 32;
    }

    // scores: two 16-key tiles, K-dim = d = 32 (single WMMA each)
#pragma unroll
    for (int j = 0; j < 2; ++j) {
      const _Float16* kp = kbase + (size_t)(j * 16 + ll) * 256 + (g ? 16 : 0);
      v16h bk = ld32B(kp);
      v8f s = {};
      s = wmma16(aq, bk, s);
#pragma unroll
      for (int r = 0; r < 8; ++r)
        sS[(r + 8 * g) * 32 + j * 16 + ll] = s[r];
    }
    __syncthreads();

    // online softmax (16 rows handled by 16 lanes)
    if (lane < 16) {
      float m_old = rowm[lane];
      float mx = m_old;
      for (int k2 = 0; k2 < 32; ++k2)
        mx = fmaxf(mx, sS[lane * 32 + k2] * scale);
      float corr = __expf(m_old - mx);
      float sum = rowsum[lane] * corr;
      for (int k2 = 0; k2 < 32; ++k2) {
        float p = __expf(sS[lane * 32 + k2] * scale - mx);
        sum += p;
        sP[lane * 32 + k2] = (_Float16)p;
      }
      rowm[lane] = mx; rowsum[lane] = sum; rowsc[lane] = corr;
    }
    __syncthreads();

    // rescale running output accumulators
#pragma unroll
    for (int r = 0; r < 8; ++r) {
      float sc = rowsc[r + 8 * g];
      acc0[r] *= sc; acc1[r] *= sc;
    }

    // P (16x32) fragment from LDS
    const _Float16* pp = &sP[ll * 32];
    v16h ap = ld32B(pp + (g ? 16 : 0));

    // O += P * V   (two d-halves)
    {
      const _Float16* vp0 = vt + (size_t)(0 * 16 + ll) * vstride + tokoff + (g ? 16 : 0);
      v16h bv0 = ld32B(vp0);
      acc0 = wmma16(ap, bv0, acc0);
      const _Float16* vp1 = vt + (size_t)(1 * 16 + ll) * vstride + tokoff + (g ? 16 : 0);
      v16h bv1 = ld32B(vp1);
      acc1 = wmma16(ap, bv1, acc1);
    }
    __syncthreads();
  }

  // normalize and write O[token][head*32 + d]
#pragma unroll
  for (int r = 0; r < 8; ++r) {
    int M = r + 8 * g;
    int token = (b * NWIN + w) * 16 + M;
    float inv = 1.0f / rowsum[M];
    Ob[(size_t)token * 256 + head * HD + ll]      = (_Float16)(acc0[r] * inv);
    Ob[(size_t)token * 256 + head * HD + 16 + ll] = (_Float16)(acc1[r] * inv);
  }
}

// ---------------------------------------------------------------------------
// Stage 5: out projection + bias + residual, scattered to BCHW f32
// grid = (128, 4), block = 128
// ---------------------------------------------------------------------------
__global__ __launch_bounds__(128) void gemm_out(
    const _Float16* __restrict__ A, const _Float16* __restrict__ Bt,
    const float* __restrict__ bias, const float* __restrict__ feat,
    float* __restrict__ out) {
  int wave = threadIdx.x >> 5;
  int lane = threadIdx.x & 31;
  int ll = lane & 15, g = lane >> 4;
  int rowbase = blockIdx.x * 64 + (wave >> 1) * 32;
  int colbase = blockIdx.y * 64 + (wave & 1) * 32;
  v8f acc[2][2] = {};
  for (int kb = 0; kb < 256; kb += 32) {
    v16h a[2], bb[2];
#pragma unroll
    for (int i = 0; i < 2; ++i) {
      const _Float16* ap = A + (size_t)(rowbase + i * 16 + ll) * 256 + kb + (g ? 16 : 0);
      a[i] = ld32B(ap);
      const _Float16* bp = Bt + (size_t)(colbase + i * 16 + ll) * 256 + kb + (g ? 16 : 0);
      bb[i] = ld32B(bp);
    }
#pragma unroll
    for (int i = 0; i < 2; ++i)
#pragma unroll
      for (int j = 0; j < 2; ++j)
        acc[i][j] = wmma16(a[i], bb[j], acc[i][j]);
  }
#pragma unroll
  for (int i = 0; i < 2; ++i)
#pragma unroll
    for (int j = 0; j < 2; ++j)
#pragma unroll
      for (int r = 0; r < 8; ++r) {
        int row = rowbase + i * 16 + r + 8 * g;     // token
        int col = colbase + j * 16 + ll;            // channel
        int b = row >> 12, w = (row >> 4) & 255, q = row & 15;
        int y = (w >> 4) * 4 + (q >> 2);
        int x = (w & 15) * 4 + (q & 3);
        size_t idx = ((size_t)(b * C_DIM + col)) * HW + y * 64 + x;
        out[idx] = acc[i][j][r] + bias[col] + feat[idx];
      }
}

// ---------------------------------------------------------------------------
extern "C" void kernel_launch(void* const* d_in, const int* in_sizes, int n_in,
                              void* d_out, int out_size, void* d_ws, size_t ws_size,
                              hipStream_t stream) {
  const float* features = (const float*)d_in[0];
  const float* qkv_w    = (const float*)d_in[1];
  const float* qkv_b    = (const float*)d_in[2];
  const float* out_w    = (const float*)d_in[3];
  const float* out_b    = (const float*)d_in[4];
  const float* ln_w     = (const float*)d_in[5];
  const float* ln_b     = (const float*)d_in[6];
  float* out = (float*)d_out;

  char* ws = (char*)d_ws;
  size_t off = 0;
  auto take = [&](size_t bytes) -> char* {
    char* p = ws + off;
    off += (bytes + 255) & ~(size_t)255;
    return p;
  };

  _Float16* wqt  = (_Float16*)take(65536 * 2);
  _Float16* wkt  = (_Float16*)take(65536 * 2);
  _Float16* wvt  = (_Float16*)take(65536 * 2);
  _Float16* wot  = (_Float16*)take(65536 * 2);
  _Float16* qtok = (_Float16*)take((size_t)8192  * 256 * 2);
  _Float16* loc  = (_Float16*)take((size_t)32768 * 256 * 2);
  _Float16* mid  = (_Float16*)take((size_t)2048  * 256 * 2);
  _Float16* glb  = (_Float16*)take((size_t)512   * 256 * 2);
  _Float16* Q16  = (_Float16*)take((size_t)8192  * 256 * 2);
  _Float16* Kl   = (_Float16*)take((size_t)32768 * 256 * 2);
  _Float16* Vl   = (_Float16*)take((size_t)32768 * 256 * 2);
  _Float16* Vlt  = (_Float16*)take((size_t)32768 * 256 * 2);
  _Float16* Km   = (_Float16*)take((size_t)2048  * 256 * 2);
  _Float16* Vm   = (_Float16*)take((size_t)2048  * 256 * 2);
  _Float16* Vmt  = (_Float16*)take((size_t)2048  * 256 * 2);
  _Float16* Kg   = (_Float16*)take((size_t)512   * 256 * 2);
  _Float16* Vg   = (_Float16*)take((size_t)512   * 256 * 2);
  _Float16* Vgt  = (_Float16*)take((size_t)512   * 256 * 2);
  _Float16* Ob   = (_Float16*)take((size_t)8192  * 256 * 2);

  convert_weights<<<256, 256, 0, stream>>>(qkv_w, out_w, wqt, wkt, wvt, wot);
  build_qtok<<<8192, 256, 0, stream>>>(features, ln_w, ln_b, qtok);
  build_loc<<<32768, 256, 0, stream>>>(features, loc);
  build_mid<<<2048, 256, 0, stream>>>(features, mid);
  build_glb<<<512, 256, 0, stream>>>(features, glb);

  gemm_f16<<<dim3(128, 4), 128, 0, stream>>>(qtok, wqt, qkv_b,       Q16);
  gemm_f16<<<dim3(512, 4), 128, 0, stream>>>(loc,  wkt, qkv_b + 256, Kl);
  gemm_f16<<<dim3(512, 4), 128, 0, stream>>>(loc,  wvt, qkv_b + 512, Vl);
  gemm_f16<<<dim3(32,  4), 128, 0, stream>>>(mid,  wkt, qkv_b + 256, Km);
  gemm_f16<<<dim3(32,  4), 128, 0, stream>>>(mid,  wvt, qkv_b + 512, Vm);
  gemm_f16<<<dim3(8,   4), 128, 0, stream>>>(glb,  wkt, qkv_b + 256, Kg);
  gemm_f16<<<dim3(8,   4), 128, 0, stream>>>(glb,  wvt, qkv_b + 512, Vg);

  transpose_local<<<32768, 256, 0, stream>>>(Vl, Vlt);
  transpose_bl<<<2048, 256, 0, stream>>>(Vm, Vmt, 1024);
  transpose_bl<<<512,  256, 0, stream>>>(Vg, Vgt, 256);

  attn_kernel<<<4096, 32, 0, stream>>>(Q16, Kl, Km, Kg, Vlt, Vmt, Vgt, Ob);

  gemm_out<<<dim3(128, 4), 128, 0, stream>>>(Ob, wot, out_b, features, out);
}
